// WaveNetModel_72103910965947
// MI455X (gfx1250) — matrix operand order repaired and checked by reading, not compiled, not run on hardware
//
#include <hip/hip_runtime.h>

#define LAYERS_ 10
#define NL_     40
#define RC_     32
#define DC_     32
#define CLS_    256
#define BB_     8
#define TT_     16384

typedef __attribute__((ext_vector_type(16))) __bf16 v16bf;
typedef __attribute__((ext_vector_type(8)))  __bf16 v8bf;
typedef __attribute__((ext_vector_type(8)))  float  v8f;
typedef __attribute__((ext_vector_type(8)))  unsigned short ushort8;
typedef __attribute__((ext_vector_type(4)))  float  f32x4;

union Frag { v16bf v; ushort8 u[2]; };

static __device__ __forceinline__ v8f wmma_bf16(v16bf a, v16bf b, v8f c) {
  return __builtin_amdgcn_wmma_f32_16x16x32_bf16(false, a, false, b, (short)0, c,
                                                 false, false);
}
// A fragment: pre-packed [frag][lane][16] bf16, contiguous 32B per lane.
static __device__ __forceinline__ Frag loadA(const __bf16* base, int frag, int lane) {
  Frag f;
  const ushort8* p = (const ushort8*)(base + ((size_t)frag * 32 + lane) * 16);
  f.u[0] = p[0]; f.u[1] = p[1];
  return f;
}
// B fragment from [t][C] bf16 activations: lane hi half holds K = hi*16 + 0..15.
static __device__ __forceinline__ Frag loadB_rows(const __bf16* rowBase, bool valid) {
  Frag f;
  if (valid) {
    const ushort8* p = (const ushort8*)rowBase;
    f.u[0] = p[0]; f.u[1] = p[1];
  } else {
    ushort8 z = {0,0,0,0,0,0,0,0};
    f.u[0] = z; f.u[1] = z;
  }
  return f;
}

// ---------------- weight pre-pack kernels (fp32 -> bf16 A fragments) ----------------
// A-fragment K map (16-bit A 16x32): K = hi*16 + j

__global__ void prepA1x1(const float* __restrict__ w, __bf16* __restrict__ dst,
                         int IC, int total) {
  int i = blockIdx.x * blockDim.x + threadIdx.x;
  if (i >= total) return;
  int j = i & 15, lane = (i >> 4) & 31;
  int KC = IC / 32;
  int kc = (i >> 9) % KC;
  int mt = i / (KC * 512);
  int hi = lane >> 4, m = mt * 16 + (lane & 15);
  int k = kc * 32 + hi * 16 + j;
  dst[i] = (__bf16)w[(size_t)m * IC + k];
}

// conv weights [NL][OC][32][2] -> frags [l][mt][tap][lane][j], K within chunk = in-channel
__global__ void prepAtap(const float* __restrict__ w, __bf16* __restrict__ dst,
                         int MT, int total) {
  int i = blockIdx.x * blockDim.x + threadIdx.x;
  if (i >= total) return;
  int j = i & 15, lane = (i >> 4) & 31, tap = (i >> 9) & 1;
  int mt = (i >> 10) % MT;
  int l  = i / (MT * 1024);
  int OC = MT * 16;
  int hi = lane >> 4, m = mt * 16 + (lane & 15);
  int k = hi * 16 + j;
  dst[i] = (__bf16)w[(((size_t)l * OC + m) * 32 + k) * 2 + tap];
}

__global__ void ksumb(const float* __restrict__ sb, float* __restrict__ dst) {
  int c = threadIdx.x;
  float s = 0.f;
  for (int l = 0; l < NL_; ++l) s += sb[l * 256 + c];
  dst[c] = s;
}

// ---------------- input 1x1 conv: x[B][256][T] fp32 -> h[B][T][32] bf16 ----------------
// 4 N-tiles (64 times) per wave.
__global__ void __launch_bounds__(32) k_in(const float* __restrict__ x,
                                           const __bf16* __restrict__ wA,
                                           const float* __restrict__ bin,
                                           __bf16* __restrict__ h) {
  int lane = threadIdx.x & 31, n = lane & 15, hi = lane >> 4;
  int tile = blockIdx.x;
  int b = tile / (TT_ / 64), tb = (tile % (TT_ / 64)) * 64;
  float bias[2][8];
#pragma unroll
  for (int mt = 0; mt < 2; ++mt)
#pragma unroll
    for (int v = 0; v < 8; ++v) bias[mt][v] = bin[mt * 16 + hi * 8 + v];
  for (int nt = 0; nt < 4; ++nt) {
    int t = tb + nt * 16 + n;
    v8f acc[2];
#pragma unroll
    for (int mt = 0; mt < 2; ++mt)
#pragma unroll
      for (int v = 0; v < 8; ++v) acc[mt][v] = bias[mt][v];
    for (int kc = 0; kc < 8; ++kc) {
      Frag bfr;
      const float* xp = x + (((size_t)b * CLS_ + kc * 32 + hi * 16) * TT_) + t;
      __bf16 tmp[16];
#pragma unroll
      for (int j = 0; j < 16; ++j) tmp[j] = (__bf16)xp[(size_t)j * TT_];
      bfr.u[0] = *(const ushort8*)&tmp[0];
      bfr.u[1] = *(const ushort8*)&tmp[8];
#pragma unroll
      for (int mt = 0; mt < 2; ++mt) {
        Frag af = loadA(wA, mt * 8 + kc, lane);
        acc[mt] = wmma_bf16(af.v, bfr.v, acc[mt]);
      }
    }
#pragma unroll
    for (int mt = 0; mt < 2; ++mt) {
      v8bf o;
#pragma unroll
      for (int v = 0; v < 8; ++v) o[v] = (__bf16)acc[mt][v];
      *(v8bf*)(h + (((size_t)b * TT_ + t) * RC_) + mt * 16 + hi * 8) = o;
    }
  }
}

// ---------------- per-layer fused filter+gate -> z = tanh(f)*sigmoid(g) ----------------
__global__ void __launch_bounds__(32) k_gate(const __bf16* __restrict__ h,
                                             const __bf16* __restrict__ fA,
                                             const __bf16* __restrict__ gA,
                                             const float* __restrict__ fb,
                                             const float* __restrict__ gb,
                                             __bf16* __restrict__ z, int d) {
  int lane = threadIdx.x & 31, n = lane & 15, hi = lane >> 4;
  int tile = blockIdx.x;
  int b = tile / (TT_ / 64), tb = (tile % (TT_ / 64)) * 64;
  // hoist the 8 weight fragments (filt/gate x 2 mtiles x 2 taps)
  Frag af[2][2], ag[2][2];
#pragma unroll
  for (int mt = 0; mt < 2; ++mt)
#pragma unroll
    for (int tap = 0; tap < 2; ++tap) {
      af[mt][tap] = loadA(fA, mt * 2 + tap, lane);
      ag[mt][tap] = loadA(gA, mt * 2 + tap, lane);
    }
  float fbv[2][8], gbv[2][8];
#pragma unroll
  for (int mt = 0; mt < 2; ++mt)
#pragma unroll
    for (int v = 0; v < 8; ++v) {
      fbv[mt][v] = fb[mt * 16 + hi * 8 + v];
      gbv[mt][v] = gb[mt * 16 + hi * 8 + v];
    }
  for (int nt = 0; nt < 4; ++nt) {
    int t = tb + nt * 16 + n;
    v8f f[2], g[2];
#pragma unroll
    for (int mt = 0; mt < 2; ++mt)
#pragma unroll
      for (int v = 0; v < 8; ++v) { f[mt][v] = fbv[mt][v]; g[mt][v] = gbv[mt][v]; }
#pragma unroll
    for (int tap = 0; tap < 2; ++tap) {
      int ts = t - ((tap == 0) ? d : 0);
      Frag bfr = loadB_rows(h + (((size_t)b * TT_ + ts) * RC_) + hi * 16, ts >= 0);
#pragma unroll
      for (int mt = 0; mt < 2; ++mt) {
        f[mt] = wmma_bf16(af[mt][tap].v, bfr.v, f[mt]);
        g[mt] = wmma_bf16(ag[mt][tap].v, bfr.v, g[mt]);
      }
    }
#pragma unroll
    for (int mt = 0; mt < 2; ++mt) {
      v8bf o;
#pragma unroll
      for (int v = 0; v < 8; ++v) {
        float zv = tanhf(f[mt][v]) * (1.f / (1.f + __expf(-g[mt][v])));
        o[v] = (__bf16)zv;
      }
      *(v8bf*)(z + (((size_t)b * TT_ + t) * DC_) + mt * 16 + hi * 8) = o;
    }
  }
}

// ---------------- per-layer residual update: h += conv(z, res_w) + res_b ----------------
__global__ void __launch_bounds__(32) k_res(const __bf16* __restrict__ z,
                                            const __bf16* __restrict__ rA,
                                            const float* __restrict__ rb,
                                            __bf16* __restrict__ h, int d) {
  int lane = threadIdx.x & 31, n = lane & 15, hi = lane >> 4;
  int tile = blockIdx.x;
  int b = tile / (TT_ / 64), tb = (tile % (TT_ / 64)) * 64;
  Frag ar[2][2];
#pragma unroll
  for (int mt = 0; mt < 2; ++mt)
#pragma unroll
    for (int tap = 0; tap < 2; ++tap) ar[mt][tap] = loadA(rA, mt * 2 + tap, lane);
  float rbv[2][8];
#pragma unroll
  for (int mt = 0; mt < 2; ++mt)
#pragma unroll
    for (int v = 0; v < 8; ++v) rbv[mt][v] = rb[mt * 16 + hi * 8 + v];
  for (int nt = 0; nt < 4; ++nt) {
    int t = tb + nt * 16 + n;
    v8f acc[2];
#pragma unroll
    for (int mt = 0; mt < 2; ++mt)
#pragma unroll
      for (int v = 0; v < 8; ++v) acc[mt][v] = rbv[mt][v];
#pragma unroll
    for (int tap = 0; tap < 2; ++tap) {
      int ts = t - ((tap == 0) ? d : 0);
      Frag bfr = loadB_rows(z + (((size_t)b * TT_ + ts) * DC_) + hi * 16, ts >= 0);
#pragma unroll
      for (int mt = 0; mt < 2; ++mt) acc[mt] = wmma_bf16(ar[mt][tap].v, bfr.v, acc[mt]);
    }
#pragma unroll
    for (int mt = 0; mt < 2; ++mt) {
      __bf16* hp = h + (((size_t)b * TT_ + t) * RC_) + mt * 16 + hi * 8;
      v8bf ho = *(const v8bf*)hp;
      v8bf o;
#pragma unroll
      for (int v = 0; v < 8; ++v) o[v] = (__bf16)((float)ho[v] + acc[mt][v]);
      *(v8bf*)hp = o;
    }
  }
}

// ------- fused end: skip-sum over all 40 layers (K=2560) + ReLU + end1 + ReLU + end2 -------
__global__ void __launch_bounds__(32) k_end(const __bf16* __restrict__ zall,
                                            const __bf16* __restrict__ sA,
                                            const float* __restrict__ ssb,
                                            const __bf16* __restrict__ e1A,
                                            const float* __restrict__ e1b,
                                            const __bf16* __restrict__ e2A,
                                            const float* __restrict__ e2b,
                                            float* __restrict__ out) {
  __shared__ __bf16 lds[16 * 256];   // 8 KB staging for C-layout -> B-layout
  int lane = threadIdx.x & 31, n = lane & 15, hi = lane >> 4;
  int tile = blockIdx.x;
  int b = tile / (TT_ / 16), t = (tile % (TT_ / 16)) * 16 + n;

  // skip accumulation: 16 M-tiles x (40 layers * 2 taps) K-chunks of 32
  v8f acc[16];
#pragma unroll
  for (int mt = 0; mt < 16; ++mt)
#pragma unroll
    for (int v = 0; v < 8; ++v) acc[mt][v] = ssb[mt * 16 + hi * 8 + v];

  for (int l = 0; l < NL_; ++l) {
    int d = 1 << (l % LAYERS_);
    const __bf16* zl = zall + (size_t)l * BB_ * TT_ * DC_;
    if (l + 1 < NL_) {   // prefetch next layer's z row (gfx1250 global_prefetch_b8)
      const __bf16* zn = zall + (size_t)(l + 1) * BB_ * TT_ * DC_;
      __builtin_prefetch(zn + (((size_t)b * TT_ + t) * DC_), 0, 0);
    }
#pragma unroll
    for (int tap = 0; tap < 2; ++tap) {
      int ts = t - ((tap == 0) ? d : 0);
      Frag bfr = loadB_rows(zl + (((size_t)b * TT_ + ts) * DC_) + hi * 16, ts >= 0);
#pragma unroll
      for (int mt = 0; mt < 16; ++mt) {
        Frag af = loadA(sA, (l * 16 + mt) * 2 + tap, lane);
        acc[mt] = wmma_bf16(af.v, bfr.v, acc[mt]);
      }
    }
  }
#pragma unroll
  for (int mt = 0; mt < 16; ++mt)
#pragma unroll
    for (int v = 0; v < 8; ++v)
      lds[n * 256 + mt * 16 + hi * 8 + v] = (__bf16)fmaxf(acc[mt][v], 0.f);
  __syncthreads();

  // end1: 256 -> 256, K = 256 from LDS
  v8f a1[16];
#pragma unroll
  for (int mt = 0; mt < 16; ++mt)
#pragma unroll
    for (int v = 0; v < 8; ++v) a1[mt][v] = e1b[mt * 16 + hi * 8 + v];
#pragma unroll
  for (int kc = 0; kc < 8; ++kc) {
    Frag bfr;
    const __bf16* rp = lds + n * 256 + kc * 32 + hi * 16;
    bfr.u[0] = *(const ushort8*)rp;
    bfr.u[1] = *(const ushort8*)(rp + 8);
#pragma unroll
    for (int mt = 0; mt < 16; ++mt) {
      Frag af = loadA(e1A, mt * 8 + kc, lane);
      a1[mt] = wmma_bf16(af.v, bfr.v, a1[mt]);
    }
  }
  __syncthreads();
#pragma unroll
  for (int mt = 0; mt < 16; ++mt)
#pragma unroll
    for (int v = 0; v < 8; ++v)
      lds[n * 256 + mt * 16 + hi * 8 + v] = (__bf16)fmaxf(a1[mt][v], 0.f);
  __syncthreads();

  // end2: 256 -> 256, fp32 result to global
  v8f a2[16];
#pragma unroll
  for (int mt = 0; mt < 16; ++mt)
#pragma unroll
    for (int v = 0; v < 8; ++v) a2[mt][v] = e2b[mt * 16 + hi * 8 + v];
#pragma unroll
  for (int kc = 0; kc < 8; ++kc) {
    Frag bfr;
    const __bf16* rp = lds + n * 256 + kc * 32 + hi * 16;
    bfr.u[0] = *(const ushort8*)rp;
    bfr.u[1] = *(const ushort8*)(rp + 8);
#pragma unroll
    for (int mt = 0; mt < 16; ++mt) {
      Frag af = loadA(e2A, mt * 8 + kc, lane);
      a2[mt] = wmma_bf16(af.v, bfr.v, a2[mt]);
    }
  }
#pragma unroll
  for (int mt = 0; mt < 16; ++mt) {
    float* op = out + (((size_t)b * TT_ + t) * CLS_) + mt * 16 + hi * 8;
    f32x4 lo, hh;
#pragma unroll
    for (int v = 0; v < 4; ++v) { lo[v] = a2[mt][v]; hh[v] = a2[mt][v + 4]; }
    *(f32x4*)op = lo;
    *(f32x4*)(op + 4) = hh;
  }
}

// ------------------------------------ host launcher ------------------------------------
extern "C" void kernel_launch(void* const* d_in, const int* in_sizes, int n_in,
                              void* d_out, int out_size, void* d_ws, size_t ws_size,
                              hipStream_t stream) {
  (void)in_sizes; (void)n_in; (void)out_size; (void)ws_size;
  const float* x      = (const float*)d_in[0];
  const float* w_in   = (const float*)d_in[1];
  const float* b_in   = (const float*)d_in[2];
  const float* filt_w = (const float*)d_in[3];
  const float* filt_b = (const float*)d_in[4];
  const float* gate_w = (const float*)d_in[5];
  const float* gate_b = (const float*)d_in[6];
  const float* res_w  = (const float*)d_in[7];
  const float* res_b  = (const float*)d_in[8];
  const float* skip_w = (const float*)d_in[9];
  const float* skip_b = (const float*)d_in[10];
  const float* e1_w   = (const float*)d_in[11];
  const float* e1_b   = (const float*)d_in[12];
  const float* e2_w   = (const float*)d_in[13];
  const float* e2_b   = (const float*)d_in[14];
  float* out = (float*)d_out;

  char* p = (char*)d_ws;
  auto carve = [&](size_t bytes) -> char* {
    char* r = p; p += (bytes + 255) & ~(size_t)255; return r;
  };
  __bf16* h     = (__bf16*)carve((size_t)BB_ * TT_ * RC_ * 2);
  __bf16* zall  = (__bf16*)carve((size_t)NL_ * BB_ * TT_ * DC_ * 2);
  __bf16* winA  = (__bf16*)carve((size_t)2 * 8 * 512 * 2);
  __bf16* filtA = (__bf16*)carve((size_t)NL_ * 2 * 2 * 512 * 2);
  __bf16* gateA = (__bf16*)carve((size_t)NL_ * 2 * 2 * 512 * 2);
  __bf16* resA  = (__bf16*)carve((size_t)NL_ * 2 * 2 * 512 * 2);
  __bf16* skipA = (__bf16*)carve((size_t)NL_ * 16 * 2 * 512 * 2);
  __bf16* e1A   = (__bf16*)carve((size_t)16 * 8 * 512 * 2);
  __bf16* e2A   = (__bf16*)carve((size_t)16 * 8 * 512 * 2);
  float* ssb    = (float*)carve(256 * 4);

  // weight pre-pack
  { int tot = 2 * 8 * 512;
    prepA1x1<<<(tot + 255) / 256, 256, 0, stream>>>(w_in, winA, 256, tot); }
  { int tot = NL_ * 2 * 2 * 512;
    prepAtap<<<(tot + 255) / 256, 256, 0, stream>>>(filt_w, filtA, 2, tot);
    prepAtap<<<(tot + 255) / 256, 256, 0, stream>>>(gate_w, gateA, 2, tot);
    prepAtap<<<(tot + 255) / 256, 256, 0, stream>>>(res_w,  resA,  2, tot); }
  { int tot = NL_ * 16 * 2 * 512;
    prepAtap<<<(tot + 255) / 256, 256, 0, stream>>>(skip_w, skipA, 16, tot); }
  { int tot = 16 * 8 * 512;
    prepA1x1<<<(tot + 255) / 256, 256, 0, stream>>>(e1_w, e1A, 256, tot);
    prepA1x1<<<(tot + 255) / 256, 256, 0, stream>>>(e2_w, e2A, 256, tot); }
  ksumb<<<1, 256, 0, stream>>>(skip_b, ssb);

  int tiles64 = BB_ * TT_ / 64;  // 2048 wave-tiles of 64 times
  int tiles16 = BB_ * TT_ / 16;  // 8192 wave-tiles of 16 times
  k_in<<<tiles64, 32, 0, stream>>>(x, winA, b_in, h);
  for (int l = 0; l < NL_; ++l) {
    int d = 1 << (l % LAYERS_);
    __bf16* zl = zall + (size_t)l * BB_ * TT_ * DC_;
    k_gate<<<tiles64, 32, 0, stream>>>(h, filtA + (size_t)l * 2048, gateA + (size_t)l * 2048,
                                       filt_b + l * 32, gate_b + l * 32, zl, d);
    k_res<<<tiles64, 32, 0, stream>>>(zl, resA + (size_t)l * 2048, res_b + l * 32, h, d);
  }
  k_end<<<tiles16, 32, 0, stream>>>(zall, skipA, ssb, e1A, e1_b, e2A, e2_b, out);
}
